// Model_71159018160808
// MI455X (gfx1250) — matrix operand order repaired and checked by reading, not compiled, not run on hardware
//
#include <hip/hip_runtime.h>

#define BB 256
#define TT 512
#define HH 64
#define NCOL 192  // 3H

typedef __attribute__((ext_vector_type(16))) __bf16 v16bf;
typedef __attribute__((ext_vector_type(8)))  float  v8f;

union Frag32 { uint4 u[2]; v16bf v; unsigned short s[16]; };
union Pack8  { unsigned short s[8]; uint4 u; };

__device__ __forceinline__ unsigned short f2bf(float f) {
  union { float f; unsigned u; } c; c.f = f;
  unsigned r = c.u + 0x7FFFu + ((c.u >> 16) & 1u);   // round-to-nearest-even
  return (unsigned short)(r >> 16);
}

__device__ __forceinline__ v8f unpack8(uint4 q) {
  v8f r;
  unsigned w[4] = {q.x, q.y, q.z, q.w};
  #pragma unroll
  for (int i = 0; i < 4; ++i) {
    union { unsigned u; float f; } lo, hi;
    lo.u = w[i] << 16; hi.u = w[i] & 0xFFFF0000u;
    r[2 * i] = lo.f; r[2 * i + 1] = hi.f;
  }
  return r;
}

// ---------------- embedding gather (fp32 -> bf16) ----------------
__global__ void k_embed(const int* __restrict__ tok, const float* __restrict__ emb,
                        unsigned short* __restrict__ x0) {
  int gid = blockIdx.x * blockDim.x + threadIdx.x;   // B*T*64
  int bt = gid >> 6, e = gid & 63;
  int tk = tok[bt];
  x0[gid] = f2bf(emb[tk * 64 + e]);
}

// ---------------- pack W[K][192] f32 -> bf16 B-fragment layout ----------------
// frag (kk,j): lane holds 16 contiguous bf16: K = kk*32 + (lane>=16)*16 + e, N = j*16 + (lane&15)
__global__ void k_packw(const float* __restrict__ w, unsigned short* __restrict__ out, int K) {
  int tid = blockIdx.x * blockDim.x + threadIdx.x;
  int total = (K >> 5) * 12 * 32;
  if (tid >= total) return;
  int lane = tid & 31, fidx = tid >> 5;
  int kk = fidx / 12, j = fidx % 12;
  int g = lane >> 4, lm = lane & 15;
  int n = j * 16 + lm;
  #pragma unroll
  for (int e = 0; e < 16; ++e)
    out[tid * 16 + e] = f2bf(w[(kk * 32 + g * 16 + e) * NCOL + n]);
}

// ---------------- input projection GEMM: xp = x @ Wi + (bi + bh[z,r]) ----------------
// One wave per (batch-tile bt, t). A fragments loaded directly from bf16 activations,
// B fragments from pre-packed weights, result stored pre-packed for the recurrence.
template<int DIN>
__global__ void k_ipgemm(const unsigned short* __restrict__ x,
                         const unsigned short* __restrict__ wp,
                         const float* __restrict__ bi,
                         const float* __restrict__ bh,
                         unsigned short* __restrict__ xp) {
  int lane = threadIdx.x & 31;
  int wv = (blockIdx.x * blockDim.x + threadIdx.x) >> 5;  // 0..16*T-1
  int bt = wv >> 9;
  int t  = wv & (TT - 1);
  int g = lane >> 4, lm = lane & 15;

  v8f acc[12];
  #pragma unroll
  for (int j = 0; j < 12; ++j) {
    int col = j * 16 + lm;
    float bv = bi[col] + (col < 2 * HH ? bh[col] : 0.f);  // fold bh for z,r gates only
    #pragma unroll
    for (int r = 0; r < 8; ++r) acc[j][r] = bv;
  }

  const unsigned short* xr = x + ((size_t)(bt * 16 + lm) * TT + t) * DIN;
  #pragma unroll
  for (int kk = 0; kk < DIN / 32; ++kk) {
    Frag32 a;
    a.u[0] = *(const uint4*)(xr + kk * 32 + g * 16);
    a.u[1] = *(const uint4*)(xr + kk * 32 + g * 16 + 8);
    const unsigned short* wk = wp + (size_t)kk * 12 * 32 * 16;
    #pragma unroll
    for (int j = 0; j < 12; ++j) {
      Frag32 b;
      const uint4* p = (const uint4*)(wk + ((j * 32 + lane) << 4));
      b.u[0] = p[0]; b.u[1] = p[1];
      acc[j] = __builtin_amdgcn_wmma_f32_16x16x32_bf16(false, a.v, false, b.v,
                                                       (short)0, acc[j], false, false);
    }
  }

  unsigned short* o = xp + ((size_t)bt * TT + t) * (12 * 32 * 8);
  #pragma unroll
  for (int j = 0; j < 12; ++j) {
    Pack8 pk;
    #pragma unroll
    for (int r = 0; r < 8; ++r) pk.s[r] = f2bf(acc[j][r]);
    *(uint4*)(o + ((j * 32 + lane) << 3)) = pk.u;
  }
}

// ---------------- GRU recurrence: one wave per 16-row batch tile, T sequential steps ----------------
__global__ void __launch_bounds__(128, 1)
k_gru(const int* __restrict__ tok,
      const unsigned short* __restrict__ xpf, const unsigned short* __restrict__ xpb,
      const unsigned short* __restrict__ whf, const unsigned short* __restrict__ whb,
      const float* __restrict__ bf_, const float* __restrict__ bb_,
      unsigned short* __restrict__ y, float* __restrict__ hfin, int write_y) {
  __shared__ uint4 whl[1536];                   // 24KB packed Wh fragments
  __shared__ unsigned short hbv[4 * 16 * 72];   // per-wave h bounce, padded rows (72)

  int dir = blockIdx.y;
  const unsigned short* xp  = dir ? xpb : xpf;
  const unsigned short* whp = dir ? whb : whf;
  const float* bvec = dir ? bb_ : bf_;          // (2,192): [0]=bi, [1]=bh

  int tid = threadIdx.x;
  {
    const uint4* src = (const uint4*)whp;
    #pragma unroll
    for (int i = 0; i < 12; ++i) whl[tid + i * 128] = src[tid + i * 128];
  }
  __syncthreads();

  int lane = tid & 31, wid = tid >> 5;
  int g = lane >> 4, lm = lane & 15;
  int bt = blockIdx.x * 4 + wid;
  int b0 = bt * 16;
  int wbase = wid * 16 * 72;

  float bhh[4];
  #pragma unroll
  for (int j = 0; j < 4; ++j) bhh[j] = bvec[192 + 128 + j * 16 + lm];  // bh_h slice

  v8f h[4];
  #pragma unroll
  for (int j = 0; j < 4; ++j)
    #pragma unroll
    for (int r = 0; r < 8; ++r) h[j][r] = 0.f;
  Frag32 hA[2];
  #pragma unroll
  for (int kk = 0; kk < 2; ++kk) {
    hA[kk].u[0] = make_uint4(0, 0, 0, 0);
    hA[kk].u[1] = make_uint4(0, 0, 0, 0);
  }

  for (int step = 0; step < TT; ++step) {
    int t = dir ? (TT - 1 - step) : step;
    const unsigned short* xpt = xp + ((size_t)bt * TT + t) * (12 * 32 * 8);

    v8f acc[12], xh[4];
    #pragma unroll
    for (int j = 0; j < 8; ++j)
      acc[j] = unpack8(*(const uint4*)(xpt + ((j * 32 + lane) << 3)));
    #pragma unroll
    for (int j = 0; j < 4; ++j) {
      xh[j] = unpack8(*(const uint4*)(xpt + (((8 + j) * 32 + lane) << 3)));
      #pragma unroll
      for (int r = 0; r < 8; ++r) acc[8 + j][r] = bhh[j];  // rec_h starts at bh_h
    }

    int tk = tok[(b0 + lm) * TT + t];
    unsigned mbits = (unsigned)__ballot(tk > 0);  // bit m = mask of batch row m

    #pragma unroll
    for (int kk = 0; kk < 2; ++kk) {
      #pragma unroll
      for (int j = 0; j < 12; ++j) {
        Frag32 w;
        const uint4* p = &whl[((kk * 12 + j) * 32 + lane) * 2];
        w.u[0] = p[0]; w.u[1] = p[1];
        acc[j] = __builtin_amdgcn_wmma_f32_16x16x32_bf16(false, hA[kk].v, false, w.v,
                                                         (short)0, acc[j], false, false);
      }
    }

    #pragma unroll
    for (int j = 0; j < 4; ++j) {
      #pragma unroll
      for (int r = 0; r < 8; ++r) {
        float z  = 1.f / (1.f + __expf(-acc[j][r]));
        float rg = 1.f / (1.f + __expf(-acc[4 + j][r]));
        float ht = xh[j][r] + rg * acc[8 + j][r];
        float hh = 2.f / (1.f + __expf(-2.f * ht)) - 1.f;   // tanh
        float hn = z * h[j][r] + (1.f - z) * hh;
        int m = r + 8 * g;
        float nv = ((mbits >> m) & 1u) ? hn : h[j][r];
        h[j][r] = nv;
        hbv[wbase + m * 72 + j * 16 + lm] = f2bf(nv);
        if (write_y)
          y[((size_t)(b0 + m) * TT + t) * 128 + dir * 64 + j * 16 + lm] = f2bf(nv);
      }
    }
    // D-layout -> A-fragment conversion through LDS (in-order within wave)
    #pragma unroll
    for (int kk = 0; kk < 2; ++kk) {
      hA[kk].u[0] = *(const uint4*)&hbv[wbase + lm * 72 + kk * 32 + g * 16];
      hA[kk].u[1] = *(const uint4*)&hbv[wbase + lm * 72 + kk * 32 + g * 16 + 8];
    }
  }

  if (!write_y) {
    #pragma unroll
    for (int j = 0; j < 4; ++j)
      #pragma unroll
      for (int r = 0; r < 8; ++r) {
        int m = r + 8 * g;
        hfin[(size_t)(b0 + m) * 128 + dir * 64 + j * 16 + lm] = h[j][r];
      }
  }
}

// ---------------- dense head: [256,128] -> 64 -> 64 -> 64 -> 28 ----------------
__global__ void k_head(const float* __restrict__ hf,
                       const float* __restrict__ wd1, const float* __restrict__ bd1,
                       const float* __restrict__ wd2, const float* __restrict__ bd2,
                       const float* __restrict__ wd3, const float* __restrict__ bd3,
                       const float* __restrict__ wo,  const float* __restrict__ bo,
                       float* __restrict__ out) {
  __shared__ float xs[128], h1[64], h2[64], h3[64];
  int b = blockIdx.x, t = threadIdx.x;
  xs[t] = hf[b * 128 + t];
  __syncthreads();
  if (t < 64) {
    float s = bd1[t];
    for (int k = 0; k < 128; ++k) s += xs[k] * wd1[k * 64 + t];
    h1[t] = fmaxf(s, 0.f);
  }
  __syncthreads();
  if (t < 64) {
    float s = bd2[t];
    for (int k = 0; k < 64; ++k) s += h1[k] * wd2[k * 64 + t];
    h2[t] = fmaxf(s, 0.f);
  }
  __syncthreads();
  if (t < 64) {
    float s = bd3[t];
    for (int k = 0; k < 64; ++k) s += h2[k] * wd3[k * 64 + t];
    h3[t] = fmaxf(s, 0.f);
  }
  __syncthreads();
  if (t < 28) {
    float s = bo[t];
    for (int k = 0; k < 64; ++k) s += h3[k] * wo[k * 28 + t];
    out[b * 28 + t] = 1.f / (1.f + __expf(-s));
  }
}

extern "C" void kernel_launch(void* const* d_in, const int* in_sizes, int n_in,
                              void* d_out, int out_size, void* d_ws, size_t ws_size,
                              hipStream_t stream) {
  const int*   tok = (const int*)d_in[0];
  const float* emb = (const float*)d_in[1];
  const float *wi[3][2], *wh[3][2], *bb[3][2];
  int idx = 2;
  for (int l = 0; l < 3; ++l)
    for (int d = 0; d < 2; ++d) {
      wi[l][d] = (const float*)d_in[idx++];
      wh[l][d] = (const float*)d_in[idx++];
      bb[l][d] = (const float*)d_in[idx++];
    }
  const float* wd1 = (const float*)d_in[20];
  const float* bd1 = (const float*)d_in[21];
  const float* wd2 = (const float*)d_in[22];
  const float* bd2 = (const float*)d_in[23];
  const float* wd3 = (const float*)d_in[24];
  const float* bd3 = (const float*)d_in[25];
  const float* wo  = (const float*)d_in[26];
  const float* bo  = (const float*)d_in[27];
  float* out = (float*)d_out;

  char* ws = (char*)d_ws;
  auto alloc = [&](size_t n) { char* p = ws; ws += (n + 255) & ~(size_t)255; return p; };

  unsigned short* x0  = (unsigned short*)alloc((size_t)BB * TT * 64 * 2);
  unsigned short* y1  = (unsigned short*)alloc((size_t)BB * TT * 128 * 2);
  unsigned short* y2  = (unsigned short*)alloc((size_t)BB * TT * 128 * 2);
  unsigned short* xpf = (unsigned short*)alloc((size_t)BB * TT * 192 * 2);
  unsigned short* xpb = (unsigned short*)alloc((size_t)BB * TT * 192 * 2);
  unsigned short *wip[3][2], *whpk[3][2];
  for (int l = 0; l < 3; ++l)
    for (int d = 0; d < 2; ++d) {
      int K = (l == 0) ? 64 : 128;
      wip[l][d]  = (unsigned short*)alloc((size_t)(K / 32) * 12 * 32 * 16 * 2);
      whpk[l][d] = (unsigned short*)alloc((size_t)2 * 12 * 32 * 16 * 2);
    }
  float* hfin = (float*)alloc((size_t)BB * 128 * 4);

  k_embed<<<(BB * TT * 64) / 256, 256, 0, stream>>>(tok, emb, x0);
  for (int l = 0; l < 3; ++l)
    for (int d = 0; d < 2; ++d) {
      int K = (l == 0) ? 64 : 128;
      int tw = (K / 32) * 12 * 32;
      k_packw<<<(tw + 127) / 128, 128, 0, stream>>>(wi[l][d], wip[l][d], K);
      k_packw<<<(2 * 12 * 32 + 127) / 128, 128, 0, stream>>>(wh[l][d], whpk[l][d], 64);
    }

  dim3 gR(4, 2);
  // layer 1 (din=64)
  k_ipgemm<64><<<2048, 128, 0, stream>>>(x0, wip[0][0], bb[0][0], bb[0][0] + 192, xpf);
  k_ipgemm<64><<<2048, 128, 0, stream>>>(x0, wip[0][1], bb[0][1], bb[0][1] + 192, xpb);
  k_gru<<<gR, 128, 0, stream>>>(tok, xpf, xpb, whpk[0][0], whpk[0][1],
                                bb[0][0], bb[0][1], y1, nullptr, 1);
  // layer 2 (din=128)
  k_ipgemm<128><<<2048, 128, 0, stream>>>(y1, wip[1][0], bb[1][0], bb[1][0] + 192, xpf);
  k_ipgemm<128><<<2048, 128, 0, stream>>>(y1, wip[1][1], bb[1][1], bb[1][1] + 192, xpb);
  k_gru<<<gR, 128, 0, stream>>>(tok, xpf, xpb, whpk[1][0], whpk[1][1],
                                bb[1][0], bb[1][1], y2, nullptr, 1);
  // layer 3 (din=128, final states only)
  k_ipgemm<128><<<2048, 128, 0, stream>>>(y2, wip[2][0], bb[2][0], bb[2][0] + 192, xpf);
  k_ipgemm<128><<<2048, 128, 0, stream>>>(y2, wip[2][1], bb[2][1], bb[2][1] + 192, xpb);
  k_gru<<<gR, 128, 0, stream>>>(tok, xpf, xpb, whpk[2][0], whpk[2][1],
                                bb[2][0], bb[2][1], nullptr, hfin, 0);

  k_head<<<BB, 128, 0, stream>>>(hfin, wd1, bd1, wd2, bd2, wd3, bd3, wo, bo, out);
}
